// MambaLayer_12713103196340
// MI455X (gfx1250) — matrix operand order repaired and checked by reading, not compiled, not run on hardware
//
#include <hip/hip_runtime.h>
#include <hip/hip_bf16.h>

// ---------------------------------------------------------------------------
// Mamba vision layer for MI455X (gfx1250). bf16 WMMA for the two large GEMMs,
// VALU for the skinny projections and the sequential selective scan.
// Compile-time shapes (template N,K + constexpr L) so all addressing folds to
// shifts/immediates; LDS-retiled f32 epilogue for coalesced scatter stores.
// ---------------------------------------------------------------------------

typedef __attribute__((ext_vector_type(16))) __bf16 v16bf;
typedef __attribute__((ext_vector_type(8)))  __bf16 v8bf;
typedef __attribute__((ext_vector_type(8)))  float  v8f;

#define DIMC     128
#define D_INNER  256
#define D_STATE  16
#define DT_RANK  8
#define NPROJ    (DT_RANK + 2 * D_STATE)   // 40
constexpr int LQ   = 128 * 128;            // 16384 tokens per image
constexpr int BSZ  = 4;
constexpr int MQ   = BSZ * LQ;             // 65536 rows

__device__ __forceinline__ __bf16 f32_to_bf16(float f) { return (__bf16)f; }
__device__ __forceinline__ float bf16_to_f32(__bf16 b) { return (float)b; }
__device__ __forceinline__ v16bf cat8(v8bf lo, v8bf hi) {
    return __builtin_shufflevector(lo, hi, 0, 1, 2, 3, 4, 5, 6, 7,
                                           8, 9, 10, 11, 12, 13, 14, 15);
}

// ---------------------------------------------------------------------------
// Weight conversion (elementwise; W_in / W_out are already (n,k) row-major,
// which is exactly the N-major B layout the GEMMs want).
// ---------------------------------------------------------------------------
__global__ void k_convert(const float* __restrict__ W_in, const float* __restrict__ W_out,
                          __bf16* __restrict__ B1, __bf16* __restrict__ B3) {
    int idx = blockIdx.x * 256 + threadIdx.x;
    if (idx < 2 * D_INNER * DIMC) {                       // 512*128
        B1[idx] = f32_to_bf16(W_in[idx]);
    } else if (idx < 2 * D_INNER * DIMC + DIMC * D_INNER) {
        int j = idx - 2 * D_INNER * DIMC;                 // 128*256
        B3[j] = f32_to_bf16(W_out[j]);
    }
}

// ---------------------------------------------------------------------------
// LayerNorm over channels; (B,128,L) channel-major -> xn bf16 (B*L, 128) rows
// ---------------------------------------------------------------------------
__global__ __launch_bounds__(256) void k_layernorm(const float* __restrict__ x,
        const float* __restrict__ w, const float* __restrict__ bp,
        __bf16* __restrict__ xn) {
    int row = blockIdx.x * 256 + threadIdx.x;     // < MQ
    int b = row >> 14, l = row & (LQ - 1);
    const float* xb = x + (size_t)b * DIMC * LQ + l;
    float s = 0.f, s2 = 0.f;
    for (int c = 0; c < DIMC; ++c) { float v = xb[(size_t)c * LQ]; s += v; s2 += v * v; }
    float mu = s * (1.f / DIMC);
    float var = s2 * (1.f / DIMC) - mu * mu;
    float rs = rsqrtf(var + 1e-5f);
    for (int c = 0; c < DIMC; ++c) {
        float v = xb[(size_t)c * LQ];
        xn[(size_t)row * DIMC + c] = f32_to_bf16((v - mu) * rs * w[c] + bp[c]);
    }
}

// ---------------------------------------------------------------------------
// WMMA GEMM core. A: (M x K) row-major bf16. Bt: (N x K) row-major bf16.
// Block tile 128(M) x 64(N), BK=32; 8 waves, each wave owns 32x32 (2x2 accs).
// ---------------------------------------------------------------------------
#define BM 128
#define BN 64
#define BK 32

struct GemmAcc { v8f a[2][2]; };

template<int N, int K>
__device__ __forceinline__ void gemm_mainloop(const __bf16* __restrict__ A,
        const __bf16* __restrict__ Bt, int m0, int n0, int tid, int wid, int lane,
        __bf16 (*As)[BK], __bf16 (*Bs)[BK], GemmAcc& g) {
    const int wm = (wid & 3) * 32, wn = (wid >> 2) * 32;
    const int arow = lane & 15, kb = (lane >> 4) * 8;
    const int bcol = lane & 15, bk = (lane >> 4) * 8;
    #pragma unroll
    for (int k0 = 0; k0 < K; k0 += BK) {
        if (k0 + BK < K)
            __builtin_prefetch(&A[(size_t)(m0 + (tid >> 1)) * K + k0 + BK], 0, 0);
        {   // stage A: 128x32, each thread copies 16 contiguous bf16 (2 x b128)
            int r = tid >> 1, c = (tid & 1) * 16;
            const __bf16* src = &A[(size_t)(m0 + r) * K + k0 + c];
            *(v8bf*)&As[r][c]     = *(const v8bf*)src;
            *(v8bf*)&As[r][c + 8] = *(const v8bf*)(src + 8);
        }
        {   // stage B (n-major): 64x32, each thread copies 8 contiguous bf16
            int r = tid >> 2, c = (tid & 3) * 8;
            *(v8bf*)&Bs[r][c] = *(const v8bf*)&Bt[(size_t)(n0 + r) * K + k0 + c];
        }
        __syncthreads();
        v16bf af[2], bfr[2];
        #pragma unroll
        for (int t = 0; t < 2; ++t) {
            v8bf alo = *(const v8bf*)&As[wm + t * 16 + arow][kb];
            v8bf ahi = *(const v8bf*)&As[wm + t * 16 + arow][16 + kb];
            af[t] = cat8(alo, ahi);
            v8bf blo = *(const v8bf*)&Bs[wn + t * 16 + bcol][bk];
            v8bf bhi = *(const v8bf*)&Bs[wn + t * 16 + bcol][16 + bk];
            bfr[t] = cat8(blo, bhi);
        }
        #pragma unroll
        for (int i = 0; i < 2; ++i)
            #pragma unroll
            for (int j = 0; j < 2; ++j)
                g.a[i][j] = __builtin_amdgcn_wmma_f32_16x16x32_bf16(
                    false, af[i], false, bfr[j], (short)0, g.a[i][j], false, false);
        __syncthreads();
    }
}

// C (M x N) row-major bf16; lanes 0..15 of each store cover 16 consecutive
// columns -> 32B coalesced runs, keep direct stores.
template<int N, int K>
__global__ __launch_bounds__(256) void k_gemm_bf16out(const __bf16* __restrict__ A,
        const __bf16* __restrict__ Bt, __bf16* __restrict__ C) {
    __shared__ __bf16 As[BM][BK];
    __shared__ __bf16 Bs[BN][BK];
    const int tid = threadIdx.x, wid = tid >> 5, lane = tid & 31;
    const int m0 = blockIdx.x * BM, n0 = blockIdx.y * BN;
    GemmAcc g = {};
    gemm_mainloop<N, K>(A, Bt, m0, n0, tid, wid, lane, As, Bs, g);
    const int wm = (wid & 3) * 32, wn = (wid >> 2) * 32;
    const int crow = (lane >> 4) * 8, ccol = lane & 15;
    #pragma unroll
    for (int i = 0; i < 2; ++i)
        #pragma unroll
        for (int j = 0; j < 2; ++j) {
            __bf16* dst = C + (size_t)(m0 + wm + i * 16 + crow) * N
                            + (n0 + wn + j * 16 + ccol);
            #pragma unroll
            for (int v = 0; v < 8; ++v)
                dst[(size_t)v * N] = f32_to_bf16(g.a[i][j][v]);
        }
}

// Final GEMM: retile through LDS, then coalesced f32 stores to (B,128,L).
template<int N, int K>
__global__ __launch_bounds__(256) void k_gemm_final(const __bf16* __restrict__ A,
        const __bf16* __restrict__ Bt, float* __restrict__ out) {
    __shared__ __bf16 As[BM][BK];
    __shared__ __bf16 Bs[BN][BK];
    __shared__ float  Cs[BN][BM + 8];     // pad 8 floats: conflict-free-ish, 32B aligned
    const int tid = threadIdx.x, wid = tid >> 5, lane = tid & 31;
    const int m0 = blockIdx.x * BM, n0 = blockIdx.y * BN;
    GemmAcc g = {};
    gemm_mainloop<N, K>(A, Bt, m0, n0, tid, wid, lane, As, Bs, g);
    const int wm = (wid & 3) * 32, wn = (wid >> 2) * 32;
    const int crow = (lane >> 4) * 8, ccol = lane & 15;
    #pragma unroll
    for (int i = 0; i < 2; ++i)
        #pragma unroll
        for (int j = 0; j < 2; ++j)
            *(v8f*)&Cs[wn + j * 16 + ccol][wm + i * 16 + crow] = g.a[i][j];
    __syncthreads();
    // block tile rows are all in one image: l = m0..m0+127 within batch b
    const int b  = m0 >> 14;              // m0 / LQ
    const int l0 = m0 & (LQ - 1);
    const int c  = tid >> 2;              // 0..63
    const int lo = (tid & 3) * 32;        // 0,32,64,96
    float* dst = out + ((size_t)(b * DIMC + n0 + c)) * LQ + l0 + lo;
    #pragma unroll
    for (int q = 0; q < 32; q += 4)
        *(float4*)(dst + q) = *(const float4*)&Cs[c][lo + q];
}

// ---------------------------------------------------------------------------
// Depthwise causal conv (k=4) + SiLU.  xz rows: [0..255]=xin, [256..511]=z
// ---------------------------------------------------------------------------
__global__ void k_conv_silu(const __bf16* __restrict__ xz, const float* __restrict__ cw,
                            const float* __restrict__ cb, __bf16* __restrict__ u) {
    long long idx = (long long)blockIdx.x * 256 + threadIdx.x;   // < MQ*256
    int d = (int)(idx & (D_INNER - 1));
    long long row = idx >> 8;                 // b*L + l
    int l = (int)(row & (LQ - 1));
    float acc = cb[d];
    #pragma unroll
    for (int j = 0; j < 4; ++j) {
        int lj = l - 3 + j;
        if (lj >= 0)
            acc += cw[d * 4 + j] * bf16_to_f32(xz[(row - 3 + j) * (2 * D_INNER) + d]);
    }
    float s = acc / (1.f + __expf(-acc));
    u[row * D_INNER + d] = f32_to_bf16(s);
}

// ---------------------------------------------------------------------------
// x_dbl = u @ W_xproj^T  (N=40, K=256)  — skinny, VALU
// ---------------------------------------------------------------------------
__global__ void k_xproj(const __bf16* __restrict__ u, const float* __restrict__ Wx,
                        float* __restrict__ xdbl, long long total) {
    long long idx = (long long)blockIdx.x * 256 + threadIdx.x;
    if (idx >= total) return;
    int o = (int)(idx % NPROJ);
    long long row = idx / NPROJ;
    const __bf16* ur = u + row * D_INNER;
    const float* wr = Wx + (size_t)o * D_INNER;
    float acc = 0.f;
    for (int k = 0; k < D_INNER; ++k) acc += bf16_to_f32(ur[k]) * wr[k];
    xdbl[idx] = acc;
}

// ---------------------------------------------------------------------------
// delta = softplus(dt @ W_dt^T + b_dt)  (K=8)
// ---------------------------------------------------------------------------
__global__ void k_delta(const float* __restrict__ xdbl, const float* __restrict__ Wdt,
                        const float* __restrict__ bdt, float* __restrict__ delta) {
    long long idx = (long long)blockIdx.x * 256 + threadIdx.x;   // < MQ*256
    int d = (int)(idx & (D_INNER - 1));
    long long row = idx >> 8;
    float acc = bdt[d];
    #pragma unroll
    for (int r = 0; r < DT_RANK; ++r) acc += xdbl[row * NPROJ + r] * Wdt[d * DT_RANK + r];
    delta[idx] = (acc > 20.f) ? acc : log1pf(__expf(acc));
}

// ---------------------------------------------------------------------------
// Selective scan (sequential over L) + gating.  One block per batch, one
// thread per channel; B_t / C_t broadcast via LDS each step.
// ---------------------------------------------------------------------------
__global__ __launch_bounds__(256) void k_scan(const float* __restrict__ delta,
        const float* __restrict__ xdbl, const __bf16* __restrict__ u,
        const __bf16* __restrict__ xz, const float* __restrict__ A_log,
        const float* __restrict__ Dsk, __bf16* __restrict__ yg) {
    int b = blockIdx.x;
    int d = threadIdx.x;                           // 0..255
    __shared__ float sB[D_STATE], sC[D_STATE];
    float negA[D_STATE];
    #pragma unroll
    for (int n = 0; n < D_STATE; ++n) negA[n] = -__expf(A_log[d * D_STATE + n]);
    float Dd = Dsk[d];
    float h[D_STATE];
    #pragma unroll
    for (int n = 0; n < D_STATE; ++n) h[n] = 0.f;
    for (int t = 0; t < LQ; ++t) {
        size_t row = (size_t)b * LQ + t;
        __syncthreads();
        if (d < 2 * D_STATE) {
            float v = xdbl[row * NPROJ + DT_RANK + d];   // [8..23]=B, [24..39]=C
            if (d < D_STATE) sB[d] = v; else sC[d - D_STATE] = v;
        }
        __syncthreads();
        float dt = delta[row * D_INNER + d];
        float ut = bf16_to_f32(u[row * D_INNER + d]);
        float du = dt * ut;
        float y = 0.f;
        #pragma unroll
        for (int n = 0; n < D_STATE; ++n) {
            float dA = __expf(dt * negA[n]);
            h[n] = dA * h[n] + du * sB[n];
            y += h[n] * sC[n];
        }
        float zt = bf16_to_f32(xz[row * (2 * D_INNER) + D_INNER + d]);
        float gate = zt / (1.f + __expf(-zt));
        yg[row * D_INNER + d] = f32_to_bf16((y + ut * Dd) * gate);
    }
}

// ---------------------------------------------------------------------------
// Launch
// ---------------------------------------------------------------------------
extern "C" void kernel_launch(void* const* d_in, const int* in_sizes, int n_in,
                              void* d_out, int out_size, void* d_ws, size_t ws_size,
                              hipStream_t stream) {
    const float* x      = (const float*)d_in[0];
    const float* ln_w   = (const float*)d_in[1];
    const float* ln_b   = (const float*)d_in[2];
    const float* W_in   = (const float*)d_in[3];
    const float* conv_w = (const float*)d_in[4];
    const float* conv_b = (const float*)d_in[5];
    const float* W_xprj = (const float*)d_in[6];
    const float* W_dt   = (const float*)d_in[7];
    const float* b_dt   = (const float*)d_in[8];
    const float* A_log  = (const float*)d_in[9];
    const float* Dskip  = (const float*)d_in[10];
    const float* W_out  = (const float*)d_in[11];
    float* out = (float*)d_out;

    char* ws = (char*)d_ws;
    auto alloc = [&](size_t bytes) {
        char* p = ws; ws += (bytes + 255) & ~(size_t)255; return p;
    };
    __bf16* xn    = (__bf16*)alloc((size_t)MQ * DIMC * 2);
    __bf16* B1    = (__bf16*)alloc((size_t)2 * D_INNER * DIMC * 2);
    __bf16* B3    = (__bf16*)alloc((size_t)DIMC * D_INNER * 2);
    __bf16* xz    = (__bf16*)alloc((size_t)MQ * 2 * D_INNER * 2);
    __bf16* u     = (__bf16*)alloc((size_t)MQ * D_INNER * 2);
    float*  xdbl  = (float*) alloc((size_t)MQ * NPROJ * 4);
    float*  delta = (float*) alloc((size_t)MQ * D_INNER * 4);
    __bf16* yg    = (__bf16*)alloc((size_t)MQ * D_INNER * 2);

    // 1. weights -> bf16 (already N-major (n,k) layout)
    {
        int total = 2 * D_INNER * DIMC + DIMC * D_INNER;
        k_convert<<<(total + 255) / 256, 256, 0, stream>>>(W_in, W_out, B1, B3);
    }
    // 2. layernorm + transpose to row-major bf16
    k_layernorm<<<MQ / 256, 256, 0, stream>>>(x, ln_w, ln_b, xn);
    // 3. xz = xn @ W_in^T   (WMMA bf16, M=65536 N=512 K=128)
    k_gemm_bf16out<2 * D_INNER, DIMC>
        <<<dim3(MQ / BM, (2 * D_INNER) / BN), 256, 0, stream>>>(xn, B1, xz);
    // 4. depthwise conv + SiLU -> u
    k_conv_silu<<<(MQ * D_INNER) / 256, 256, 0, stream>>>(xz, conv_w, conv_b, u);
    // 5. x_dbl = u @ W_xproj^T
    {
        long long total = (long long)MQ * NPROJ;
        k_xproj<<<(unsigned)((total + 255) / 256), 256, 0, stream>>>(u, W_xprj, xdbl, total);
    }
    // 6. delta = softplus(dt @ W_dt^T + b_dt)
    k_delta<<<(MQ * D_INNER) / 256, 256, 0, stream>>>(xdbl, W_dt, b_dt, delta);
    // 7. selective scan + gating -> yg (bf16)
    k_scan<<<BSZ, D_INNER, 0, stream>>>(delta, xdbl, u, xz, A_log, Dskip, yg);
    // 8. out = yg @ W_out^T, coalesced scatter to (B,128,L)  (WMMA bf16)
    k_gemm_final<DIMC, D_INNER>
        <<<dim3(MQ / BM, DIMC / BN), 256, 0, stream>>>(yg, B3, out);
}
